// LSTM_cell_31361851195863
// MI455X (gfx1250) — matrix operand order repaired and checked by reading, not compiled, not run on hardware
//
#include <hip/hip_runtime.h>

#define B_TOTAL 131072
#define D_IN    256
#define H_DIM   256
#define K_TOT   512
#define N_TOT   1024
#define BM      64                      /* batch rows per block (4 M-tiles) */
#define THREADS 512                     /* 16 wave32 */
#define LDSA_STRIDE 520                 /* bf16 per A row: 512 + 8 pad (16B aligned, bank-spread) */
#define LDS_BYTES (BM * LDSA_STRIDE * 2)  /* 66560 B */

typedef __attribute__((ext_vector_type(16))) __bf16 v16bf;
typedef __attribute__((ext_vector_type(8)))  float  v8f;

union BFrag { uint4 u[2]; v16bf v; };

__device__ __forceinline__ unsigned int f2bf(float f) {
  unsigned int u = __float_as_uint(f);
  u += 0x7FFFu + ((u >> 16) & 1u);   // round-to-nearest-even
  return u >> 16;
}

__device__ __forceinline__ float sigmoidf_fast(float v) {
  return 1.0f / (1.0f + __expf(-v));
}

// Pack W[n][k] = (k<256 ? wx : wh)[n][k%256] as bf16 pairs, row-major [1024][512].
__global__ void prep_weights(const float* __restrict__ wx,
                             const float* __restrict__ wh,
                             unsigned int* __restrict__ Wp) {
  int idx  = blockIdx.x * blockDim.x + threadIdx.x;  // 0 .. 1024*256-1 (bf16 pairs)
  int n    = idx >> 8;
  int k    = (idx & 255) * 2;
  const float* src = (k < D_IN) ? (wx + (size_t)n * D_IN + k)
                                : (wh + (size_t)n * H_DIM + (k - D_IN));
  Wp[idx] = f2bf(src[0]) | (f2bf(src[1]) << 16);
}

__global__ __launch_bounds__(THREADS, 1)
void lstm_wmma(const float* __restrict__ x,
               const float* __restrict__ ht,
               const float* __restrict__ ct,
               const unsigned short* __restrict__ Wbf,
               const float* __restrict__ bx,
               const float* __restrict__ bh,
               float* __restrict__ out_h,
               float* __restrict__ out_c) {
  extern __shared__ char smem[];
  unsigned short* Albs = (unsigned short*)smem;  // [BM][LDSA_STRIDE] bf16

  const int  tid = threadIdx.x;
  const long b0  = (long)blockIdx.x * BM;

  // ---- Stage A = [x | ht] rows b0..b0+63 into LDS as bf16 (coalesced b128 reads) ----
  for (int i = 0; i < (BM * 128) / THREADS; ++i) {
    int flat = tid + i * THREADS;        // row-major float4-chunks, 128 per row
    int r    = flat >> 7;
    int q    = flat & 127;
    float4 v = (q < 64) ? ((const float4*)(x  + (b0 + r) * D_IN))[q]
                        : ((const float4*)(ht + (b0 + r) * H_DIM))[q - 64];
    uint2 pk;
    pk.x = f2bf(v.x) | (f2bf(v.y) << 16);
    pk.y = f2bf(v.z) | (f2bf(v.w) << 16);
    *(uint2*)(Albs + r * LDSA_STRIDE + q * 4) = pk;   // ds_store_b64
  }
  __syncthreads();

  const int wave  = tid >> 5;      // 16 wave32 per block
  const int lane  = tid & 31;
  const int laneN = lane & 15;
  const int laneH = lane >> 4;     // half-wave select
  const int h0    = wave * 16;     // this wave's 16 hidden columns

  // N-tile t4 = gate t4 (i,f,o,g) of hidden columns h0..h0+15:  n = t4*256 + h0 + laneN
  v8f acc[4][4] = {};              // [M-tile][gate]

  const unsigned short* Ar[4];
#pragma unroll
  for (int m = 0; m < 4; ++m)
    Ar[m] = Albs + (m * 16 + laneN) * LDSA_STRIDE;
  const unsigned short* wr[4];
#pragma unroll
  for (int t4 = 0; t4 < 4; ++t4)
    wr[t4] = Wbf + (size_t)(t4 * H_DIM + h0 + laneN) * K_TOT;

  // ---- K loop: 16 steps of K=32; 4 A frags + 4 B frags -> 16 WMMAs per step ----
  for (int ks = 0; ks < 16; ++ks) {
    // A fragment per ISA layout: lanes<16 -> K {ka..+7, ka+16..+23}; lanes>=16 offset +8
    int ka = ks * 32 + laneH * 8;
    BFrag a[4];
#pragma unroll
    for (int m = 0; m < 4; ++m) {
      a[m].u[0] = *(const uint4*)(Ar[m] + ka);
      a[m].u[1] = *(const uint4*)(Ar[m] + ka + 16);
    }
    // B fragment: same K order as A: lanes<16 -> K {kb..+7, kb+16..+23}; lanes>=16 offset +8 (B[k][n] = W[n][k])
    int kb = ks * 32 + laneH * 8;
#pragma unroll
    for (int t4 = 0; t4 < 4; ++t4) {
      BFrag b;
      b.u[0] = *(const uint4*)(wr[t4] + kb);
      b.u[1] = *(const uint4*)(wr[t4] + kb + 16);
#pragma unroll
      for (int m = 0; m < 4; ++m)
        acc[m][t4] = __builtin_amdgcn_wmma_f32_16x16x32_bf16(
            false, a[m].v, false, b.v, (short)0, acc[m][t4], false, false);
    }
  }

  // ---- Register-resident LSTM epilogue (each lane owns column h, rows of its C tiles) ----
  const int h  = h0 + laneN;
  const float bi = bx[h]             + bh[h];
  const float bf = bx[H_DIM + h]     + bh[H_DIM + h];
  const float bo = bx[2 * H_DIM + h] + bh[2 * H_DIM + h];
  const float bg = bx[3 * H_DIM + h] + bh[3 * H_DIM + h];

#pragma unroll
  for (int m = 0; m < 4; ++m) {
    v8f pi = acc[m][0], pf = acc[m][1], po = acc[m][2], pg = acc[m][3];
#pragma unroll
    for (int vv = 0; vv < 8; ++vv) {
      int  row = m * 16 + laneH * 8 + vv;   // C layout: VGPR v -> M = v + 8*(lane>=16)
      long b   = b0 + row;
      float gi = sigmoidf_fast(pi[vv] + bi);
      float gf = sigmoidf_fast(pf[vv] + bf);
      float go = sigmoidf_fast(po[vv] + bo);
      float gg = sigmoidf_fast(pg[vv] + bg);   // reference: sigmoid on g too
      float c  = gf * ct[b * H_DIM + h] + gi * gg;
      float ac = fabsf(c);
      float e  = __expf(-2.0f * ac);
      float th = __builtin_copysignf((1.0f - e) / (1.0f + e), c);
      out_h[b * H_DIM + h] = go * th;
      out_c[b * H_DIM + h] = c;
    }
  }
}

extern "C" void kernel_launch(void* const* d_in, const int* in_sizes, int n_in,
                              void* d_out, int out_size, void* d_ws, size_t ws_size,
                              hipStream_t stream) {
  (void)in_sizes; (void)n_in; (void)out_size; (void)ws_size;
  const float* x  = (const float*)d_in[0];
  const float* ht = (const float*)d_in[1];
  const float* ct = (const float*)d_in[2];
  const float* wx = (const float*)d_in[3];
  const float* bx = (const float*)d_in[4];
  const float* wh = (const float*)d_in[5];
  const float* bh = (const float*)d_in[6];

  unsigned int* Wp = (unsigned int*)d_ws;          // 1 MB packed bf16 weights
  float* out_h = (float*)d_out;
  float* out_c = out_h + (size_t)B_TOTAL * H_DIM;  // (h, c) concatenated

  prep_weights<<<(N_TOT * (K_TOT / 2)) / 256, 256, 0, stream>>>(wx, wh, Wp);
  lstm_wmma<<<B_TOTAL / BM, THREADS, LDS_BYTES, stream>>>(
      x, ht, ct, (const unsigned short*)d_ws, bx, bh, out_h, out_c);
}
